// RNN_8899172237927
// MI455X (gfx1250) — matrix operand order repaired and checked by reading, not compiled, not run on hardware
//
#include <hip/hip_runtime.h>
#include <hip/hip_bf16.h>
#include <math.h>

// Problem dims (compile-time constants from the reference)
#define I_DIM  1024
#define H_DIM  1024
#define B_DIM  64
#define S_DIM  512
#define M_DIM  (B_DIM * S_DIM)   // 32768 rows for the x-projection GEMM
#define NT_W   64                // N tiles in W (1024/16)
#define KT_WX  32                // K tiles of the Wx part (1024/32)
#define KT_ALL 64                // K tiles of full W (2048/32)
#define WH_TILE_OFF ((size_t)KT_WX * NT_W * 32 * 16)  // packed-elem offset of Wh region
#define RECUR_BLOCKS 32          // persistent blocks in the recurrence kernel

typedef __bf16 bf16_t;
typedef __attribute__((ext_vector_type(16))) __bf16 v16bf;
typedef __attribute__((ext_vector_type(8)))  __bf16 v8bf;
typedef __attribute__((ext_vector_type(4)))  __bf16 v4bf;
typedef __attribute__((ext_vector_type(8)))  float  v8f;

// ---------------------------------------------------------------------------
// A-operand loader: 16x32 bf16 tile, row-major source with row stride lda.
// Contiguous k order (matches packed B): lane m(0..15) holds K=0..15,
// lane m+16 holds K=16..31 -> two contiguous 16B loads per lane.
// ---------------------------------------------------------------------------
__device__ inline v16bf load_a16x32(const bf16_t* __restrict__ A, int lda, int lane) {
  const int m  = lane & 15;
  const int kb = (lane & 16) ? 16 : 0;
  const v8bf lo = *(const v8bf*)(A + (size_t)m * lda + kb);
  const v8bf hi = *(const v8bf*)(A + (size_t)m * lda + kb + 8);
  v16bf r;
#pragma unroll
  for (int i = 0; i < 8; ++i) { r[i] = lo[i]; r[i + 8] = hi[i]; }
  return r;
}

// ---------------------------------------------------------------------------
// Device-wide sense-reversing barrier (co-resident persistent blocks).
// bar[0] = arrival counter, bar[1] = generation.
// ---------------------------------------------------------------------------
__device__ inline void grid_barrier(unsigned* __restrict__ bar, unsigned nblk) {
  __syncthreads();
  if (threadIdx.x == 0) {
    const unsigned g = __hip_atomic_load(bar + 1, __ATOMIC_ACQUIRE, __HIP_MEMORY_SCOPE_AGENT);
    const unsigned prev = __hip_atomic_fetch_add(bar, 1u, __ATOMIC_ACQ_REL, __HIP_MEMORY_SCOPE_AGENT);
    if (prev == nblk - 1u) {
      __hip_atomic_store(bar, 0u, __ATOMIC_RELAXED, __HIP_MEMORY_SCOPE_AGENT);
      __hip_atomic_store(bar + 1, g + 1u, __ATOMIC_RELEASE, __HIP_MEMORY_SCOPE_AGENT);
    } else {
      while (__hip_atomic_load(bar + 1, __ATOMIC_ACQUIRE, __HIP_MEMORY_SCOPE_AGENT) == g)
        __builtin_amdgcn_s_sleep(1);
    }
  }
  __syncthreads();
}

// ---------------------------------------------------------------------------
// Utility kernels
// ---------------------------------------------------------------------------
__global__ void k_zero_barrier(unsigned* p) { p[0] = 0u; p[1] = 0u; }

__global__ void k_f32_to_bf16(const float* __restrict__ src, bf16_t* __restrict__ dst, int n4) {
  const int i = blockIdx.x * blockDim.x + threadIdx.x;
  if (i >= n4) return;
  const float4 v = ((const float4*)src)[i];
  v4bf o;
  o[0] = (bf16_t)v.x; o[1] = (bf16_t)v.y; o[2] = (bf16_t)v.z; o[3] = (bf16_t)v.w;
  ((v4bf*)dst)[i] = o;
}

// Pre-swizzle W (2048x1024 f32, row-major K x N) into the WMMA B-operand
// layout: [kt(64)][nt(64)][lane(32)][16 bf16], so the GEMM hot loop does one
// contiguous 32-byte load per lane. Contiguous k order (same as load_a16x32): lanes 0-15 hold
// K=0..15, lanes 16-31 hold K=16..31, column = lane&15.
__global__ void k_pack_w(const float* __restrict__ W, bf16_t* __restrict__ Wp) {
  const int t    = blockIdx.x * blockDim.x + threadIdx.x;   // 0 .. 64*64*32-1
  const int lane = t & 31;
  const int nt   = (t >> 5) & 63;
  const int kt   = t >> 11;
  const int n    = nt * 16 + (lane & 15);
  const int kb   = kt * 32 + ((lane & 16) ? 16 : 0);
  v16bf o;
#pragma unroll
  for (int e = 0; e < 16; ++e)
    o[e] = (bf16_t)W[(size_t)(kb + e) * H_DIM + n];
  *(v16bf*)(Wp + (size_t)t * 16) = o;
}

// ---------------------------------------------------------------------------
// x-projection GEMM: out[M=32768, N=1024] = A[M,K=1024] (bf16) x Wx (packed).
// grid (512, 32), block 128 (4 waves). Each wave computes TWO 16x16 N-tiles
// sharing one A operand -> two independent v_wmma accumulator chains per wave
// (hides the WMMA->WMMA hazard slots) and half the A load traffic per wmma.
// ---------------------------------------------------------------------------
__global__ __launch_bounds__(128) void k_xproj_gemm(const bf16_t* __restrict__ A,
                                                    const bf16_t* __restrict__ Wp,
                                                    float* __restrict__ out) {
  const int lane = threadIdx.x & 31;
  const int wave = threadIdx.x >> 5;
  const int mt   = blockIdx.x * 4 + wave;   // 0..2047
  const int nt0  = blockIdx.y * 2;          // even n-tile
  const bf16_t* Arow = A + (size_t)mt * 16 * I_DIM;

  v8f c0 = {}, c1 = {};
#pragma unroll 4
  for (int kt = 0; kt < KT_WX; ++kt) {
    const v16bf a  = load_a16x32(Arow + kt * 32, I_DIM, lane);
    const v16bf b0 = *(const v16bf*)(Wp + (((size_t)kt * NT_W + nt0)     * 32 + lane) * 16);
    const v16bf b1 = *(const v16bf*)(Wp + (((size_t)kt * NT_W + nt0 + 1) * 32 + lane) * 16);
    if (kt + 1 < KT_WX)
      __builtin_prefetch(Wp + (((size_t)(kt + 1) * NT_W + nt0) * 32 + lane) * 16, 0, 1);
    c0 = __builtin_amdgcn_wmma_f32_16x16x32_bf16(false, a, false, b0, (short)0, c0, false, false);
    c1 = __builtin_amdgcn_wmma_f32_16x16x32_bf16(false, a, false, b1, (short)0, c1, false, false);
  }

  const int n0    = nt0 * 16 + (lane & 15);
  const int mbase = mt * 16 + ((lane & 16) ? 8 : 0);
#pragma unroll
  for (int r = 0; r < 8; ++r) {
    float* row = out + (size_t)(mbase + r) * H_DIM;
    row[n0]      = c0[r];
    row[n0 + 16] = c1[r];
  }
}

// ---------------------------------------------------------------------------
// Persistent recurrence kernel: 32 blocks x 128 threads. Block b owns columns
// [32*b, 32*b+32) (two N-tiles); its 4 waves cover batch rows 0..63, each wave
// running two independent wmma chains. Per step:
//   c = h_prev @ Wh (tiles), h = tanh(c + xproj[s] + b), ping-pong h, barrier.
// ---------------------------------------------------------------------------
template <bool OUT_BF16>
__global__ __launch_bounds__(128) void k_recur(const float*  __restrict__ xproj,   // (B,S,H) f32
                                               const bf16_t* __restrict__ WhP,     // packed Wh tiles
                                               const float*  __restrict__ bias,    // (H)
                                               const float*  __restrict__ h0,      // (B,H) f32
                                               bf16_t*       __restrict__ hbuf,    // 2*B*H bf16 ping-pong
                                               bf16_t*       __restrict__ out_bf,  // (B,S,H) bf16 or null
                                               float*        __restrict__ out_f,   // (B,S,H) f32 or null
                                               float*        __restrict__ hid_out, // (B,H) f32
                                               unsigned*     __restrict__ bar) {
  const int lane = threadIdx.x & 31;
  const int wave = threadIdx.x >> 5;     // m-tile (batch rows wave*16..+16)
  const int nt0  = blockIdx.x * 2;       // even n-tile owned by this block
  const int n0   = nt0 * 16 + (lane & 15);
  bf16_t* hb0 = hbuf;
  bf16_t* hb1 = hbuf + (size_t)B_DIM * H_DIM;

  // Prologue: convert h0 (f32) -> hb0 (bf16) cooperatively.
  for (int i = blockIdx.x * blockDim.x + threadIdx.x; i < B_DIM * H_DIM;
       i += blockDim.x * gridDim.x)
    hb0[i] = (bf16_t)h0[i];
  __threadfence();
  grid_barrier(bar, RECUR_BLOCKS);

  const float bn0 = bias[n0];
  const float bn1 = bias[n0 + 16];

  for (int s = 0; s < S_DIM; ++s) {
    const bf16_t* hin  = (s & 1) ? hb1 : hb0;
    bf16_t*       hout = (s & 1) ? hb0 : hb1;
    const bf16_t* Ar   = hin + (size_t)wave * 16 * H_DIM;

    v8f c0 = {}, c1 = {};
#pragma unroll 4
    for (int kt = 0; kt < 32; ++kt) {   // Wh is rows 1024..2047 -> 32 k-tiles
      const v16bf a  = load_a16x32(Ar + kt * 32, H_DIM, lane);
      const v16bf b0 = *(const v16bf*)(WhP + (((size_t)kt * NT_W + nt0)     * 32 + lane) * 16);
      const v16bf b1 = *(const v16bf*)(WhP + (((size_t)kt * NT_W + nt0 + 1) * 32 + lane) * 16);
      c0 = __builtin_amdgcn_wmma_f32_16x16x32_bf16(false, a, false, b0, (short)0, c0, false, false);
      c1 = __builtin_amdgcn_wmma_f32_16x16x32_bf16(false, a, false, b1, (short)0, c1, false, false);
    }

    const int mbase = wave * 16 + ((lane & 16) ? 8 : 0);
#pragma unroll
    for (int r = 0; r < 8; ++r) {
      const int m = mbase + r;
      const size_t oidx = ((size_t)m * S_DIM + s) * H_DIM + n0;  // tile 0; tile 1 at +16
      float v0 = c0[r] + xproj[oidx]      + bn0;
      float v1 = c1[r] + xproj[oidx + 16] + bn1;
      v0 = tanhf(v0);
      v1 = tanhf(v1);
      hout[(size_t)m * H_DIM + n0]      = (bf16_t)v0;
      hout[(size_t)m * H_DIM + n0 + 16] = (bf16_t)v1;
      if (OUT_BF16) {
        out_bf[oidx]      = (bf16_t)v0;
        out_bf[oidx + 16] = (bf16_t)v1;
      } else {
        out_f[oidx]      = v0;
        out_f[oidx + 16] = v1;
      }
      if (s == S_DIM - 1) {
        hid_out[(size_t)m * H_DIM + n0]      = v0;
        hid_out[(size_t)m * H_DIM + n0 + 16] = v1;
      }
    }
    __threadfence();
    grid_barrier(bar, RECUR_BLOCKS);
  }
}

// ---------------------------------------------------------------------------
// Host launcher
// ---------------------------------------------------------------------------
extern "C" void kernel_launch(void* const* d_in, const int* in_sizes, int n_in,
                              void* d_out, int out_size, void* d_ws, size_t ws_size,
                              hipStream_t stream) {
  (void)in_sizes; (void)n_in; (void)out_size; (void)ws_size;

  const float* x      = (const float*)d_in[0];           // (B,S,I)
  const float* hidden = (const float*)d_in[1];           // (L,B,H)
  const float* W0     = (const float*)d_in[2];           // (I+H, H)
  const float* b0     = (const float*)d_in[3];           // (1,H)
  const float* W1     = (const float*)d_in[4];           // (2H, H)
  const float* b1     = (const float*)d_in[5];           // (1,H)

  float* out1 = (float*)d_out;                                    // (B,S,H)
  float* hid0 = out1 + (size_t)B_DIM * S_DIM * H_DIM;             // (B,H)
  float* hid1 = hid0 + (size_t)B_DIM * H_DIM;                     // (B,H)

  // Workspace carve-up (bf16 region first, then f32, then barrier state).
  const size_t WP_ELEMS = (size_t)2048 * 1024;   // packed elems per W
  bf16_t* Wp0     = (bf16_t*)d_ws;
  bf16_t* Wp1     = Wp0 + WP_ELEMS;
  bf16_t* x_bf    = Wp1 + WP_ELEMS;                               // M*I
  bf16_t* out0_bf = x_bf + (size_t)M_DIM * I_DIM;                 // M*H
  bf16_t* hbuf    = out0_bf + (size_t)M_DIM * H_DIM;              // 2*B*H
  float*  xproj   = (float*)(hbuf + (size_t)2 * B_DIM * H_DIM);   // M*H f32
  unsigned* bar   = (unsigned*)(xproj + (size_t)M_DIM * H_DIM);   // 2 u32

  // 0) reset barrier state (ws is poisoned with 0xAA before timing).
  k_zero_barrier<<<1, 1, 0, stream>>>(bar);

  // 1) activations f32 -> bf16 (vectorized x4).
  {
    const int n4 = (M_DIM * I_DIM) / 4;
    k_f32_to_bf16<<<(n4 + 255) / 256, 256, 0, stream>>>(x, x_bf, n4);
  }

  // 2) pre-swizzle W0, W1 into WMMA B-operand tile layout (64*64*32 lanes).
  k_pack_w<<<(KT_ALL * NT_W * 32) / 256, 256, 0, stream>>>(W0, Wp0);
  k_pack_w<<<(KT_ALL * NT_W * 32) / 256, 256, 0, stream>>>(W1, Wp1);

  // 3) layer 0 input projection: xproj = x @ Wx0.
  k_xproj_gemm<<<dim3(M_DIM / 64, NT_W / 2), 128, 0, stream>>>(x_bf, Wp0, xproj);

  // 4) layer 0 recurrence: ys -> out0_bf (bf16, feeds layer 1), hT -> hid0.
  k_recur<true><<<RECUR_BLOCKS, 128, 0, stream>>>(xproj, Wp0 + WH_TILE_OFF, b0, hidden,
                                                  hbuf, out0_bf, nullptr, hid0, bar);

  // 5) layer 1 input projection: xproj = out0 @ Wx1 (xproj buffer reused).
  k_xproj_gemm<<<dim3(M_DIM / 64, NT_W / 2), 128, 0, stream>>>(out0_bf, Wp1, xproj);

  // 6) layer 1 recurrence: ys -> d_out (f32), hT -> hid1.
  k_recur<false><<<RECUR_BLOCKS, 128, 0, stream>>>(xproj, Wp1 + WH_TILE_OFF, b1,
                                                   hidden + (size_t)B_DIM * H_DIM,
                                                   hbuf, nullptr, out1, hid1, bar);
}
